// CustomAttention_29643864277218
// MI455X (gfx1250) — matrix operand order repaired and checked by reading, not compiled, not run on hardware
//
#include <hip/hip_runtime.h>

// ---------------------------------------------------------------------------
// MI455X (gfx1250) flash attention, bf16 WMMA (v_wmma_f32_16x16x32_bf16),
// async-to-LDS double-buffered K/V staging (ASYNCcnt path).
// ---------------------------------------------------------------------------

#define EMBED   384
#define HEADS   6
#define HDIM    64
#define BATCH   8
#define SEQ     2048
#define M_TOTAL (BATCH * SEQ)   // 16384 rows
#define QKV_OUT (3 * EMBED)     // 1152

typedef __attribute__((ext_vector_type(16))) __bf16       v16bf;
typedef __attribute__((ext_vector_type(8)))  float        v8f;
typedef __attribute__((ext_vector_type(4)))  unsigned int u32x4;
typedef __attribute__((ext_vector_type(4)))  int          i32x4;

union Frag {
    v16bf  v;
    u32x4  q[2];
    __bf16 e[16];
};

__device__ __forceinline__ v8f wmma_bf16(const Frag& a, const Frag& b, v8f c) {
    // 8 args: (neg_a, A, neg_b, B, c_mod, C, reuse_a, reuse_b)
    return __builtin_amdgcn_wmma_f32_16x16x32_bf16(
        false, a.v, false, b.v, (short)0, c, false, false);
}

// ---- CDNA5 async-to-LDS copy (ASYNCcnt) -----------------------------------
// clang signature (from diagnostic): (v4i AS1* src, v4i AS3* dst, Ii, Ii)
#if __has_builtin(__builtin_amdgcn_global_load_async_to_lds_b128)
#define ASYNC_COPY_B128(gp, lp)                                               \
    __builtin_amdgcn_global_load_async_to_lds_b128(                           \
        (__attribute__((address_space(1))) i32x4*)(gp),                       \
        (__attribute__((address_space(3))) i32x4*)(lp), 0, 0)
#else
#define ASYNC_COPY_B128(gp, lp)                                               \
    asm volatile("global_load_async_to_lds_b128 %0, %1, off"                  \
                 :: "v"((unsigned)(uintptr_t)(lp)), "v"(gp) : "memory")
#endif

#if __has_builtin(__builtin_amdgcn_s_wait_asynccnt)
#define WAIT_ASYNC(n) __builtin_amdgcn_s_wait_asynccnt(n)
#else
#define WAIT_ASYNC(n) asm volatile("s_wait_asynccnt " #n ::: "memory")
#endif

// ---------------------------------------------------------------------------
// Weight transpose + f32 -> bf16 convert: w[rows][cols] -> wt[cols][rows]
// (so GEMM B-operands read contiguous K per output column)
// ---------------------------------------------------------------------------
__global__ void k_transpose_bf16(const float* __restrict__ w,
                                 __bf16* __restrict__ wt,
                                 int rows, int cols) {
    int idx = blockIdx.x * blockDim.x + threadIdx.x;
    if (idx >= rows * cols) return;
    int r = idx / cols, c = idx - r * cols;
    wt[c * rows + r] = (__bf16)w[idx];
}

// ---------------------------------------------------------------------------
// QKV projection: qkv = x @ qkv_w + b, split into Q [BH][N][D], K [BH][N][D],
// V stored transposed [BH][D][N] (so PV's B-operand is K-contiguous).
// One wave computes a 16x64 output tile; K-loop over 384 in steps of 32.
// ---------------------------------------------------------------------------
__global__ __launch_bounds__(32) void k_qkv(const float*  __restrict__ x,
                                            const __bf16* __restrict__ wT,
                                            const float*  __restrict__ bias,
                                            __bf16* __restrict__ Q,
                                            __bf16* __restrict__ K,
                                            __bf16* __restrict__ VT) {
    const int lane = threadIdx.x & 31;
    const int lo = lane & 15, hi = lane >> 4;
    const int m0 = blockIdx.y * 16;
    const int n0 = blockIdx.x * 64;

    v8f acc[4] = {};
    const float* xrow = x + (size_t)(m0 + lo) * EMBED;

    for (int k0 = 0; k0 < EMBED; k0 += 32) {
        // A fragment (16x32 bf16): lane = row, contiguous K chunk at hi*16
        Frag a;
        const int ka = k0 + hi * 16;
#pragma unroll
        for (int i = 0; i < 8; ++i) a.e[i]     = (__bf16)xrow[ka + i];
#pragma unroll
        for (int i = 0; i < 8; ++i) a.e[8 + i] = (__bf16)xrow[ka + 8 + i];

#pragma unroll
        for (int t = 0; t < 4; ++t) {
            // B fragment (32x16): lane = col, contiguous K chunk at hi*16
            Frag b;
            const __bf16* bp = wT + (size_t)(n0 + t * 16 + lo) * EMBED + k0 + hi * 16;
            b.q[0] = ((const u32x4*)bp)[0];
            b.q[1] = ((const u32x4*)bp)[1];
            acc[t] = wmma_bf16(a, b, acc[t]);
        }
    }

    // scatter with bias; D-layout: vgpr r -> row r (lanes 0-15) / r+8 (16-31)
#pragma unroll
    for (int t = 0; t < 4; ++t) {
        const int ncol = n0 + t * 16 + lo;       // [0, 1152)
        const float bv = bias[ncol];
        const int s   = ncol / EMBED;            // 0=Q 1=K 2=V (uniform per tile)
        const int rem = ncol - s * EMBED;
        const int h   = rem >> 6;
        const int d   = rem & 63;
#pragma unroll
        for (int r = 0; r < 8; ++r) {
            const int m  = m0 + r + hi * 8;
            const int b_ = m >> 11;              // / SEQ
            const int ns = m & (SEQ - 1);
            const int bh = b_ * HEADS + h;
            const __bf16 val = (__bf16)(acc[t][r] + bv);
            if (s == 0)      Q [((size_t)bh * SEQ + ns) * HDIM + d] = val;
            else if (s == 1) K [((size_t)bh * SEQ + ns) * HDIM + d] = val;
            else             VT[((size_t)bh * HDIM + d) * SEQ + ns] = val;
        }
    }
}

// ---------------------------------------------------------------------------
// Flash attention: 4 waves / workgroup, each wave owns a 16-row Q tile
// (64 rows per WG, same (b,h)). A 64-key K/V block is staged into LDS with
// double-buffered global_load_async_to_lds_b128 and shared by all 4 waves.
// Per block per wave: 8 wmma (S = Q Kt), online softmax (shfl row-reduce),
// P via LDS (C-layout -> A-layout), 8 wmma (O += P V).
// ---------------------------------------------------------------------------
#define KV_TILE   (64 * 64)      // elems per K (or V) buffer
#define P_BASE    (4 * KV_TILE)  // P region start (after K dbl-buf + V dbl-buf)

__global__ __launch_bounds__(128) void k_flash(const __bf16* __restrict__ Q,
                                               const __bf16* __restrict__ K,
                                               const __bf16* __restrict__ VT,
                                               __bf16* __restrict__ O) {
    // LDS carve-up: K dbl-buf (2x8KB) | V dbl-buf (2x8KB) | P per wave (4x2KB)
    __shared__ __bf16 smem[P_BASE + 4 * 16 * 64];

    const int tid  = threadIdx.x;
    const int wave = tid >> 5;
    const int lane = tid & 31;
    const int lo = lane & 15, hi = lane >> 4;

    __bf16* Pw = smem + P_BASE + wave * (16 * 64);

    const int m0 = blockIdx.x * 64 + wave * 16;
    const int h  = blockIdx.y;
    const int b  = blockIdx.z;
    const int bh = b * HEADS + h;

    const __bf16* Qb = Q  + (size_t)bh * SEQ * HDIM;
    const __bf16* Kb = K  + (size_t)bh * SEQ * HDIM;
    const __bf16* Vb = VT + (size_t)bh * HDIM * SEQ;

    // cooperative async stage of one 64-key K + V block (8 x b128 per thread)
    // buffer select by integer offset (no LDS-pointer aggregates: lld can't
    // relocate addrspacecast constant initializers)
    auto stage = [&](int buf, int nb) {
        __bf16* kdst = smem + buf * KV_TILE;
        __bf16* vdst = smem + 2 * KV_TILE + buf * KV_TILE;
#pragma unroll
        for (int i = 0; i < 4; ++i) {
            const int chunk = tid + i * 128;      // 0..511
            const int row   = chunk >> 3;         // key row
            const int c8    = (chunk & 7) * 8;    // elem offset in row
            ASYNC_COPY_B128(Kb + (size_t)(nb + row) * HDIM + c8,
                            kdst + chunk * 8);
        }
#pragma unroll
        for (int i = 0; i < 4; ++i) {
            const int chunk = tid + i * 128;
            const int row   = chunk >> 3;         // d row of V^T
            const int c8    = (chunk & 7) * 8;
            ASYNC_COPY_B128(Vb + (size_t)row * SEQ + nb + c8,
                            vdst + chunk * 8);
        }
    };

    // Q fragments for d = 0..31 and 32..63 (A layout)
    Frag qf[2];
    {
        const __bf16* qrow = Qb + (size_t)(m0 + lo) * HDIM;
#pragma unroll
        for (int kk = 0; kk < 2; ++kk) {
            const int d0 = kk * 32 + hi * 16;
            qf[kk].q[0] = ((const u32x4*)(qrow + d0))[0];
            qf[kk].q[1] = ((const u32x4*)(qrow + d0 + 8))[0];
        }
    }

    float mrun[8], lrun[8];
    v8f acc[4] = {};
#pragma unroll
    for (int r = 0; r < 8; ++r) { mrun[r] = -3.0e38f; lrun[r] = 0.0f; }

    const float SCALE = 0.125f;                 // 1/sqrt(64)
    const float LOG2E = 1.4426950408889634f;

    stage(0, 0);
    int cur = 0;
    for (int nb = 0; nb < SEQ; nb += 64) {
        const bool has_next = (nb + 64 < SEQ);
        if (has_next) {
            stage(cur ^ 1, nb + 64);            // prefetch next block
            WAIT_ASYNC(0x8);                    // retire current block's 8/thread
        } else {
            WAIT_ASYNC(0x0);
        }
        __syncthreads();                        // K/V tile visible to all waves

        const __bf16* Kt = smem + cur * KV_TILE;
        const __bf16* Vt = smem + 2 * KV_TILE + cur * KV_TILE;

        // ---- S = Q Kt (16x64), scaled -----------------------------------
        float sf[4][8];
#pragma unroll
        for (int j = 0; j < 4; ++j) {
            const __bf16* krow = Kt + (j * 16 + lo) * HDIM;
            Frag kf0, kf1;                      // B layout: contiguous K chunk
            kf0.q[0] = ((const u32x4*)(krow + hi * 16))[0];
            kf0.q[1] = ((const u32x4*)(krow + hi * 16))[1];
            kf1.q[0] = ((const u32x4*)(krow + 32 + hi * 16))[0];
            kf1.q[1] = ((const u32x4*)(krow + 32 + hi * 16))[1];
            v8f s = {};
            s = wmma_bf16(qf[0], kf0, s);
            s = wmma_bf16(qf[1], kf1, s);
#pragma unroll
            for (int r = 0; r < 8; ++r) sf[j][r] = s[r] * SCALE;
        }

        // ---- online softmax (rows live in 16-lane halves) ---------------
        float alpha[8];
#pragma unroll
        for (int r = 0; r < 8; ++r) {
            float rm = fmaxf(fmaxf(sf[0][r], sf[1][r]),
                             fmaxf(sf[2][r], sf[3][r]));
#pragma unroll
            for (int off = 1; off < 16; off <<= 1)
                rm = fmaxf(rm, __shfl_xor(rm, off, 32));
            const float mnew = fmaxf(mrun[r], rm);
            alpha[r] = exp2f((mrun[r] - mnew) * LOG2E);
            mrun[r]  = mnew;
            float rs = 0.0f;
#pragma unroll
            for (int j = 0; j < 4; ++j) {
                const float p = exp2f((sf[j][r] - mnew) * LOG2E);
                sf[j][r] = p;
                rs += p;
            }
#pragma unroll
            for (int off = 1; off < 16; off <<= 1)
                rs += __shfl_xor(rs, off, 32);
            lrun[r] = lrun[r] * alpha[r] + rs;
        }
#pragma unroll
        for (int t = 0; t < 4; ++t)
#pragma unroll
            for (int r = 0; r < 8; ++r) acc[t][r] *= alpha[r];

        // ---- P: C-layout -> LDS row-major [16][64] -> A-layout ----------
#pragma unroll
        for (int j = 0; j < 4; ++j)
#pragma unroll
            for (int r = 0; r < 8; ++r)
                Pw[(r + hi * 8) * 64 + j * 16 + lo] = (__bf16)sf[j][r];
        __syncthreads();

        Frag pf[2];
#pragma unroll
        for (int kk = 0; kk < 2; ++kk) {
            const __bf16* prow = Pw + lo * 64 + kk * 32 + hi * 16;
            pf[kk].q[0] = ((const u32x4*)prow)[0];
            pf[kk].q[1] = ((const u32x4*)(prow + 8))[0];
        }

        // ---- O += P V  (V tile is [d][n]: B-operand K-contiguous) -------
#pragma unroll
        for (int t = 0; t < 4; ++t) {
#pragma unroll
            for (int kk = 0; kk < 2; ++kk) {
                const __bf16* vrow = Vt + (t * 16 + lo) * 64 + kk * 32 + hi * 16;
                Frag vf;
                vf.q[0] = ((const u32x4*)vrow)[0];
                vf.q[1] = ((const u32x4*)vrow)[1];
                acc[t] = wmma_bf16(pf[kk], vf, acc[t]);
            }
        }
        __syncthreads();     // all reads of K/V/P done before restage/rewrite
        cur ^= 1;
    }

    // ---- normalize + store to [B][N][C] bf16 ----------------------------
#pragma unroll
    for (int r = 0; r < 8; ++r) {
        const float inv = 1.0f / lrun[r];
        const int m = m0 + r + hi * 8;
#pragma unroll
        for (int t = 0; t < 4; ++t)
            O[((size_t)(b * SEQ + m)) * EMBED + h * HDIM + t * 16 + lo] =
                (__bf16)(acc[t][r] * inv);
    }
}

// ---------------------------------------------------------------------------
// Output projection: out = AO @ proj_w + b  (bf16 WMMA, f32 output)
// ---------------------------------------------------------------------------
__global__ __launch_bounds__(32) void k_proj(const __bf16* __restrict__ A,
                                             const __bf16* __restrict__ pT,
                                             const float*  __restrict__ bias,
                                             float* __restrict__ out) {
    const int lane = threadIdx.x & 31;
    const int lo = lane & 15, hi = lane >> 4;
    const int m0 = blockIdx.y * 16;
    const int n0 = blockIdx.x * 64;

    v8f acc[4] = {};
    const __bf16* arow = A + (size_t)(m0 + lo) * EMBED;

    for (int k0 = 0; k0 < EMBED; k0 += 32) {
        Frag a;
        a.q[0] = ((const u32x4*)(arow + k0 + hi * 16))[0];
        a.q[1] = ((const u32x4*)(arow + k0 + hi * 16 + 8))[0];
#pragma unroll
        for (int t = 0; t < 4; ++t) {
            Frag b;
            const __bf16* bp = pT + (size_t)(n0 + t * 16 + lo) * EMBED + k0 + hi * 16;
            b.q[0] = ((const u32x4*)bp)[0];
            b.q[1] = ((const u32x4*)bp)[1];
            acc[t] = wmma_bf16(a, b, acc[t]);
        }
    }

#pragma unroll
    for (int t = 0; t < 4; ++t) {
        const int ncol = n0 + t * 16 + lo;
        const float bv = bias[ncol];
#pragma unroll
        for (int r = 0; r < 8; ++r) {
            const int m = m0 + r + hi * 8;
            out[(size_t)m * EMBED + ncol] = acc[t][r] + bv;
        }
    }
}

// ---------------------------------------------------------------------------
extern "C" void kernel_launch(void* const* d_in, const int* in_sizes, int n_in,
                              void* d_out, int out_size, void* d_ws, size_t ws_size,
                              hipStream_t stream) {
    const float* x      = (const float*)d_in[0];
    const float* qkv_w  = (const float*)d_in[1];
    const float* qkv_b  = (const float*)d_in[2];
    const float* proj_w = (const float*)d_in[3];
    const float* proj_b = (const float*)d_in[4];
    float* out = (float*)d_out;

    // workspace carve-up (bf16), ~52 MB total
    __bf16* wT = (__bf16*)d_ws;                                  // [1152][384]
    __bf16* pT = wT + (size_t)QKV_OUT * EMBED;                   // [384][384]
    __bf16* Qb = pT + (size_t)EMBED * EMBED;                     // [BH][N][D]
    __bf16* Kb = Qb + (size_t)BATCH * HEADS * SEQ * HDIM;        // [BH][N][D]
    __bf16* VT = Kb + (size_t)BATCH * HEADS * SEQ * HDIM;        // [BH][D][N]
    __bf16* AO = VT + (size_t)BATCH * HEADS * SEQ * HDIM;        // [B*N][C]

    {
        int n = EMBED * QKV_OUT;
        k_transpose_bf16<<<(n + 255) / 256, 256, 0, stream>>>(qkv_w, wT, EMBED, QKV_OUT);
    }
    {
        int n = EMBED * EMBED;
        k_transpose_bf16<<<(n + 255) / 256, 256, 0, stream>>>(proj_w, pT, EMBED, EMBED);
    }
    k_qkv  <<<dim3(QKV_OUT / 64, M_TOTAL / 16), 32, 0, stream>>>(x, wT, qkv_b, Qb, Kb, VT);
    k_flash<<<dim3(SEQ / 64, HEADS, BATCH),    128, 0, stream>>>(Qb, Kb, VT, AO);
    k_proj <<<dim3(EMBED / 64, M_TOTAL / 16),   32, 0, stream>>>(AO, pT, proj_b, out);
}
